// MiniLLM_46626164965582
// MI455X (gfx1250) — matrix operand order repaired and checked by reading, not compiled, not run on hardware
//
#include <hip/hip_runtime.h>
#include <math.h>

// ---------------- sizes (match reference) ----------------
#define Vv 32000
#define Dd 1024
#define Hh 16
#define Ll 4
#define Bb 2
#define Ss 2048
#define TOK (Bb * Ss)   // 4096
#define HDc 64

typedef __attribute__((ext_vector_type(16))) __bf16 v16bf;
typedef __attribute__((ext_vector_type(8)))  __bf16 bf16x8;
typedef __attribute__((ext_vector_type(2)))  __bf16 bf16x2;
typedef __attribute__((ext_vector_type(8)))  float  v8f;
typedef __attribute__((ext_vector_type(4)))  float  f32x4;

__device__ __forceinline__ v8f wmma_bf16(v16bf a, v16bf b, v8f c) {
  return __builtin_amdgcn_wmma_f32_16x16x32_bf16(
      false, a, false, b, (short)0, c, false, false);
}

__device__ __forceinline__ v16bf cat8(bf16x8 lo, bf16x8 hi) {
  return __builtin_shufflevector(lo, hi, 0, 1, 2, 3, 4, 5, 6, 7,
                                         8, 9, 10, 11, 12, 13, 14, 15);
}

__device__ __forceinline__ float gelu_exact(float x) {
  return 0.5f * x * (1.0f + erff(x * 0.70710678118654752f));
}

__device__ __forceinline__ unsigned lds_off(const void* p) {
  return (unsigned)(uintptr_t)p;   // generic LDS ptr: low 32 bits = LDS byte offset
}

// ---------------- embedding: x = tok_emb[ids] + pos_emb ----------------
__global__ __launch_bounds__(256) void embed_kernel(
    const int* __restrict__ ids, const float* __restrict__ tok,
    const float* __restrict__ pos, float* __restrict__ x) {
  int token = blockIdx.x;
  int s = token % Ss;
  int id = ids[token];
  int t = threadIdx.x;
  f32x4 a = ((const f32x4*)(tok + (size_t)id * Dd))[t];
  f32x4 b = ((const f32x4*)(pos + (size_t)s * Dd))[t];
  ((f32x4*)(x + (size_t)token * Dd))[t] = a + b;
}

// ---------------- layernorm (block per token, 1 float4/thread) ----------------
__global__ __launch_bounds__(256) void ln_kernel(
    const float* __restrict__ x, const float* __restrict__ g,
    const float* __restrict__ b, float* __restrict__ o) {
  int token = blockIdx.x, t = threadIdx.x;
  f32x4 v = ((const f32x4*)(x + (size_t)token * Dd))[t];
  float s  = v.x + v.y + v.z + v.w;
  float s2 = v.x * v.x + v.y * v.y + v.z * v.z + v.w * v.w;
  __shared__ float rs[256], rq[256];
  rs[t] = s; rq[t] = s2;
  __syncthreads();
  for (int off = 128; off; off >>= 1) {
    if (t < off) { rs[t] += rs[t + off]; rq[t] += rq[t + off]; }
    __syncthreads();
  }
  float mu  = rs[0] * (1.0f / Dd);
  float var = rq[0] * (1.0f / Dd) - mu * mu;
  float inv = rsqrtf(var + 1e-5f);
  f32x4 gg = ((const f32x4*)g)[t];
  f32x4 bb = ((const f32x4*)b)[t];
  f32x4 r;
  r.x = (v.x - mu) * inv * gg.x + bb.x;
  r.y = (v.y - mu) * inv * gg.y + bb.y;
  r.z = (v.z - mu) * inv * gg.z + bb.z;
  r.w = (v.w - mu) * inv * gg.w + bb.w;
  ((f32x4*)(o + (size_t)token * Dd))[t] = r;
}

// ---------------- double-buffered bf16 WMMA GEMM: C = epi(A[M,K] @ B[K,N]) ---
// Block tile 128x64, 8 waves, 32x32 per wave (4 WMMAs per wave per K-step).
// A tile: async-DMA (global_load_async_to_lds_b128) into fp32 LDS buffers.
// B tile: 2(k)x4(n) patches -> v_cvt_pk_bf16_f32 pairs -> transposed ds_store_b32.
// EPI (compile-time): 0 = none, 2 = gelu(+bias), 3 = +bias + resid
#define BM 128
#define BN 64
#define BK 32
template <int EPI>
__global__ __launch_bounds__(256) void gemm_nn(
    const float* __restrict__ A, const float* __restrict__ B,
    const float* __restrict__ bias, const float* __restrict__ resid,
    float* __restrict__ C, int M, int N, int K) {
  __shared__ f32x4  sA[2][BM][BK / 4];    // 2 x 16 KB
  __shared__ bf16x8 sBt[2][BN][BK / 8];   // 2 x 4 KB

  const int tid  = threadIdx.x;
  const int lane = tid & 31, wid = tid >> 5;
  const int wm = wid & 3, wn = wid >> 2;          // 4 x 2 waves of 32x32
  const int bm = blockIdx.y * BM, bn = blockIdx.x * BN;
  const int half = lane >> 4, l16 = lane & 15;

  v8f acc[2][2] = {{{}, {}}, {{}, {}}};

  // A staging: 128 rows x 32 k fp32 -> 16 floats/thread = 4 async b128
  const int arow = tid >> 1, acol = (tid & 1) * 16;
  // B staging: 2(k) x 4(n) patch per thread
  const int bk2 = (tid >> 4) * 2;      // even k: 0..30
  const int bn4 = (tid & 15) * 4;      // n: 0..60

  const float* apRow  = A + (size_t)(bm + arow) * K + acol;
  const float* bpBase = B + bn + bn4;

  // NOTE: all LDS addrspacecasts are computed per-use (locals), never stored in
  // arrays -- static initializers with addrspacecast(3->0) break ld.lld.
#define STAGE_A(buf, kk)                                                  \
  {                                                                       \
    unsigned la_ = lds_off(&sA[buf][arow][acol >> 2]);                    \
    asm volatile(                                                         \
        "global_load_async_to_lds_b128 %0, %1, off\n\t"                   \
        "global_load_async_to_lds_b128 %0, %1, off offset:16\n\t"         \
        "global_load_async_to_lds_b128 %0, %1, off offset:32\n\t"         \
        "global_load_async_to_lds_b128 %0, %1, off offset:48"             \
        :: "v"(la_),                                                      \
           "v"((unsigned long long)(uintptr_t)(apRow + (kk)))             \
        : "memory");                                                      \
  }

#define STAGE_B(buf, kk)                                                  \
  {                                                                       \
    __bf16* sb_ = (__bf16*)&sBt[buf][0][0];                               \
    const float* bp_ = bpBase + (size_t)((kk) + bk2) * N;                 \
    f32x4 w0 = *(const f32x4*)bp_;        /* row k   */                   \
    f32x4 w1 = *(const f32x4*)(bp_ + N);  /* row k+1 */                   \
    _Pragma("unroll")                                                     \
    for (int j = 0; j < 4; ++j) {                                         \
      bf16x2 pk_;                                                         \
      pk_[0] = (__bf16)w0[j];                                             \
      pk_[1] = (__bf16)w1[j];                                             \
      *(bf16x2*)(sb_ + (bn4 + j) * BK + bk2) = pk_;                       \
    }                                                                     \
  }

  // ---- prologue: fill buffer 0 ----
  STAGE_A(0, 0);
  STAGE_B(0, 0);
  asm volatile("s_wait_asynccnt 0x0" ::: "memory");
  __syncthreads();

  const int nk = K / BK;
  for (int i = 0; i < nk; ++i) {
    const int p = i & 1, q = p ^ 1;
    if (i + 1 < nk) {            // overlap next tile's DMA/loads with compute
      STAGE_A(q, (i + 1) * BK);
      STAGE_B(q, (i + 1) * BK);
    }

    // ---- compute buffer p: 2 A-frags x 2 B-frags -> 4 WMMAs ----
    v16bf af[2], bf[2];
    #pragma unroll
    for (int mi = 0; mi < 2; ++mi) {
      const int am = wm * 32 + mi * 16 + l16;
      f32x4 a0 = sA[p][am][2 * half];
      f32x4 a1 = sA[p][am][2 * half + 1];
      f32x4 a2 = sA[p][am][4 + 2 * half];
      f32x4 a3 = sA[p][am][5 + 2 * half];
      #pragma unroll
      for (int j = 0; j < 4; ++j) {
        af[mi][j]      = (__bf16)a0[j];
        af[mi][4 + j]  = (__bf16)a1[j];
        af[mi][8 + j]  = (__bf16)a2[j];
        af[mi][12 + j] = (__bf16)a3[j];
      }
    }
    #pragma unroll
    for (int ni = 0; ni < 2; ++ni) {
      const int n0 = wn * 32 + ni * 16 + l16;
      bf[ni] = cat8(sBt[p][n0][half], sBt[p][n0][2 + half]);
    }
    #pragma unroll
    for (int mi = 0; mi < 2; ++mi)
      #pragma unroll
      for (int ni = 0; ni < 2; ++ni)
        acc[mi][ni] = wmma_bf16(af[mi], bf[ni], acc[mi][ni]);

    if (i + 1 < nk) {
      asm volatile("s_wait_asynccnt 0x0" ::: "memory");
      __syncthreads();           // single barrier per K-step
    }
  }

  // ---- epilogue (straight-line; EPI is compile-time) ----
  #pragma unroll
  for (int mi = 0; mi < 2; ++mi)
    #pragma unroll
    for (int ni = 0; ni < 2; ++ni) {
      const int col = bn + wn * 32 + ni * 16 + l16;
      const float bv = (EPI >= 1) ? bias[col] : 0.0f;
      #pragma unroll
      for (int r = 0; r < 8; ++r) {
        int row = bm + wm * 32 + mi * 16 + r + 8 * half;
        float v0 = acc[mi][ni][r] + bv;
        if (EPI == 2) v0 = gelu_exact(v0);
        size_t idx = (size_t)row * N + col;
        if (EPI == 3) v0 += resid[idx];
        C[idx] = v0;
      }
    }
#undef STAGE_A
#undef STAGE_B
}

// ---------------- flash attention: 1 wave per (b, h, 16-query tile) ----------------
__global__ __launch_bounds__(32) void attn_kernel(
    const float* __restrict__ Q, const float* __restrict__ Km,
    const float* __restrict__ Vm, float* __restrict__ O) {
  __shared__ bf16x8 sP[16][4];          // 16x32 P tile, bf16
  const int lane = threadIdx.x;
  const int half = lane >> 4, l16 = lane & 15;
  const int nqt = Ss / 16;
  const int qt = blockIdx.x % nqt;
  const int hh = (blockIdx.x / nqt) % Hh;
  const int bb = blockIdx.x / (nqt * Hh);

  // Q fragments for the two 32-wide k-steps covering HD=64
  v16bf qf[2];
  const int qrow = qt * 16 + l16;
  const float* qp = Q + (size_t)(bb * Ss + qrow) * Dd + hh * HDc;
  #pragma unroll
  for (int s = 0; s < 2; ++s)
    #pragma unroll
    for (int e = 0; e < 16; ++e) {
      int kk = s * 32 + half * 16 + e;
      qf[s][e] = (__bf16)qp[kk];
    }

  v8f acc[4] = {{}, {}, {}, {}};
  float mrow[8], srow[8];
  #pragma unroll
  for (int r = 0; r < 8; ++r) { mrow[r] = -1e30f; srow[r] = 0.f; }
  const float scale = 0.125f;   // 1/sqrt(64)
  const int qmax = qt * 16 + 15;

  for (int kb = 0; kb <= qmax; kb += 32) {
    // ---- scores: 16q x 32k via 2 k-steps x 2 key tiles ----
    v8f sc0 = {}, sc1 = {};
    #pragma unroll
    for (int s = 0; s < 2; ++s) {
      v16bf k0v, k1v;
      const float* kp0 = Km + (size_t)(bb * Ss + kb + l16)      * Dd + hh * HDc + s * 32 + half * 16;
      const float* kp1 = Km + (size_t)(bb * Ss + kb + 16 + l16) * Dd + hh * HDc + s * 32 + half * 16;
      #pragma unroll
      for (int e = 0; e < 16; ++e) { k0v[e] = (__bf16)kp0[e]; k1v[e] = (__bf16)kp1[e]; }
      sc0 = wmma_bf16(qf[s], k0v, sc0);
      sc1 = wmma_bf16(qf[s], k1v, sc1);
    }
    // ---- online softmax update ----
    #pragma unroll
    for (int r = 0; r < 8; ++r) {
      int qr = qt * 16 + r + 8 * half;
      int key0 = kb + l16, key1 = key0 + 16;
      float s0 = (key0 <= qr) ? sc0[r] * scale : -1e30f;
      float s1 = (key1 <= qr) ? sc1[r] * scale : -1e30f;
      float tm = fmaxf(s0, s1);
      #pragma unroll
      for (int m2 = 8; m2; m2 >>= 1) tm = fmaxf(tm, __shfl_xor(tm, m2, 32));
      float nm   = fmaxf(mrow[r], tm);
      float corr = __expf(mrow[r] - nm);
      float p0 = __expf(s0 - nm), p1 = __expf(s1 - nm);
      float ps = p0 + p1;
      #pragma unroll
      for (int m2 = 8; m2; m2 >>= 1) ps += __shfl_xor(ps, m2, 32);
      srow[r] = srow[r] * corr + ps;
      mrow[r] = nm;
      #pragma unroll
      for (int t = 0; t < 4; ++t) acc[t][r] *= corr;
      __bf16* sPf = (__bf16*)&sP[0][0];
      sPf[(r + 8 * half) * 32 + l16]      = (__bf16)p0;   // C-layout -> LDS
      sPf[(r + 8 * half) * 32 + 16 + l16] = (__bf16)p1;
    }
    asm volatile("s_wait_dscnt 0x0" ::: "memory");  // LDS RAW: stores -> A-frag reads
    v16bf pa = cat8(sP[l16][2 * half], sP[l16][2 * half + 1]);  // 2 x ds_load_b128

    // ---- ctx += P @ V (4 d-tiles of 16) ----
    #pragma unroll
    for (int t = 0; t < 4; ++t) {
      v16bf bv;
      #pragma unroll
      for (int e = 0; e < 16; ++e) {
        int key = kb + half * 16 + e;
        bv[e] = (__bf16)Vm[(size_t)(bb * Ss + key) * Dd + hh * HDc + t * 16 + l16];
      }
      acc[t] = wmma_bf16(pa, bv, acc[t]);
    }
  }

  #pragma unroll
  for (int r = 0; r < 8; ++r) {
    float inv = 1.f / srow[r];
    int token = bb * Ss + qt * 16 + r + 8 * half;
    #pragma unroll
    for (int t = 0; t < 4; ++t)
      O[(size_t)token * Dd + hh * HDc + t * 16 + l16] = acc[t][r] * inv;
  }
}

// ---------------- lm-head: logits = A[M,K] @ Bt[N,K]^T (tied embeddings) ----------------
__global__ __launch_bounds__(256) void gemm_bt(
    const float* __restrict__ A, const float* __restrict__ Bt,
    float* __restrict__ C, int M, int N, int K) {
  const int lane = threadIdx.x & 31, wid = threadIdx.x >> 5;
  const int half = lane >> 4, l16 = lane & 15;
  const int n0 = blockIdx.x * 128 + wid * 16;
  const int m0 = blockIdx.y * 16;
  v8f acc = {};
  const float* ap = A  + (size_t)(m0 + l16) * K;
  const float* bp = Bt + (size_t)(n0 + l16) * K;
  for (int k0 = 0; k0 < K; k0 += 32) {
    v16bf a, b;
    #pragma unroll
    for (int e = 0; e < 16; ++e) {
      int ka = (e < 8) ? (half * 8 + e) : (16 + half * 8 + (e - 8));
      a[e] = (__bf16)ap[k0 + half * 16 + e];
      b[e] = (__bf16)bp[k0 + half * 16 + e];   // contiguous 16 floats per lane
    }
    acc = wmma_bf16(a, b, acc);
  }
  #pragma unroll
  for (int r = 0; r < 8; ++r)
    C[(size_t)(m0 + r + 8 * half) * N + n0 + l16] = acc[r];
}

// ---------------- loss: mean over tokens of (logsumexp(row) - row[target]) ----------------
__global__ __launch_bounds__(256) void loss_kernel(
    const float* __restrict__ logits, const int* __restrict__ targets,
    float* __restrict__ out_loss) {
  int token = blockIdx.x, tid = threadIdx.x;
  const float* row = logits + (size_t)token * Vv;
  const f32x4* row4 = (const f32x4*)row;
  float m = -1e30f, s = 0.f;
  for (int j = tid; j < Vv / 4; j += 256) {
    f32x4 xv = row4[j];
    #pragma unroll
    for (int c = 0; c < 4; ++c) {
      float x = xv[c];
      if (x > m) { s = s * __expf(m - x) + 1.f; m = x; }
      else       { s += __expf(x - m); }
    }
  }
  __shared__ float sm[256], ssv[256];
  sm[tid] = m; ssv[tid] = s;
  __syncthreads();
  for (int off = 128; off; off >>= 1) {
    if (tid < off) {
      float m2 = sm[tid + off], s2 = ssv[tid + off];
      float mn = fmaxf(sm[tid], m2);
      ssv[tid] = ssv[tid] * __expf(sm[tid] - mn) + s2 * __expf(m2 - mn);
      sm[tid]  = mn;
    }
    __syncthreads();
  }
  if (tid == 0) {
    float lse = sm[0] + __logf(ssv[0]);
    float lt  = row[targets[token]];
    atomicAdd(out_loss, (lse - lt) * (1.0f / (float)TOK));
  }
}

// ---------------- launcher ----------------
extern "C" void kernel_launch(void* const* d_in, const int* in_sizes, int n_in,
                              void* d_out, int out_size, void* d_ws, size_t ws_size,
                              hipStream_t stream) {
  const int*   ids  = (const int*)  d_in[0];
  const int*   tgt  = (const int*)  d_in[1];
  const float* tok  = (const float*)d_in[2];
  const float* pos  = (const float*)d_in[3];
  const float* Wq   = (const float*)d_in[4];
  const float* Wk   = (const float*)d_in[5];
  const float* Wv   = (const float*)d_in[6];
  const float* Wp   = (const float*)d_in[7];
  const float* bp   = (const float*)d_in[8];
  const float* ln1g = (const float*)d_in[9];
  const float* ln1b = (const float*)d_in[10];
  const float* ln2g = (const float*)d_in[11];
  const float* ln2b = (const float*)d_in[12];
  const float* W1   = (const float*)d_in[13];
  const float* b1   = (const float*)d_in[14];
  const float* W2   = (const float*)d_in[15];
  const float* b2   = (const float*)d_in[16];
  const float* lnfg = (const float*)d_in[17];
  const float* lnfb = (const float*)d_in[18];

  float* out = (float*)d_out;                    // logits [TOK*Vv] then loss [1]

  float* x   = (float*)d_ws;                     // [TOK, D]
  float* h   = x   + (size_t)TOK * Dd;           // [TOK, D]
  float* q   = h   + (size_t)TOK * Dd;           // [TOK, D]
  float* k   = q   + (size_t)TOK * Dd;
  float* v   = k   + (size_t)TOK * Dd;
  float* ctx = v   + (size_t)TOK * Dd;
  float* h1  = q;                                // [TOK, 4D] aliases q..ctx (free by then)

  embed_kernel<<<TOK, 256, 0, stream>>>(ids, tok, pos, x);

  const dim3 gD (Dd / BN,     TOK / BM);   // N=1024
  const dim3 g4D(4 * Dd / BN, TOK / BM);   // N=4096

  for (int i = 0; i < Ll; ++i) {
    const size_t wo = (size_t)i * Dd * Dd;
    ln_kernel<<<TOK, 256, 0, stream>>>(x, ln1g + i * Dd, ln1b + i * Dd, h);
    gemm_nn<0><<<gD, 256, 0, stream>>>(h, Wq + wo, nullptr, nullptr, q, TOK, Dd, Dd);
    gemm_nn<0><<<gD, 256, 0, stream>>>(h, Wk + wo, nullptr, nullptr, k, TOK, Dd, Dd);
    gemm_nn<0><<<gD, 256, 0, stream>>>(h, Wv + wo, nullptr, nullptr, v, TOK, Dd, Dd);
    attn_kernel<<<Bb * Hh * (Ss / 16), 32, 0, stream>>>(q, k, v, ctx);
    gemm_nn<3><<<gD, 256, 0, stream>>>(ctx, Wp + wo, bp + i * Dd, x, x, TOK, Dd, Dd);
    ln_kernel<<<TOK, 256, 0, stream>>>(x, ln2g + i * Dd, ln2b + i * Dd, h);
    gemm_nn<2><<<g4D, 256, 0, stream>>>(h, W1 + (size_t)i * Dd * 4 * Dd, b1 + (size_t)i * 4 * Dd,
                                        nullptr, h1, TOK, 4 * Dd, Dd);
    gemm_nn<3><<<gD, 256, 0, stream>>>(h1, W2 + (size_t)i * 4 * Dd * Dd, b2 + i * Dd,
                                       x, x, TOK, Dd, 4 * Dd);
  }

  ln_kernel<<<TOK, 256, 0, stream>>>(x, lnfg, lnfb, h);
  gemm_bt<<<dim3(Vv / 128, TOK / 16), 256, 0, stream>>>(h, tok, out, TOK, Vv, Dd);

  (void)hipMemsetAsync(out + (size_t)TOK * Vv, 0, sizeof(float), stream);
  loss_kernel<<<TOK, 256, 0, stream>>>(out, tgt, out + (size_t)TOK * Vv);
}
